// SimpleGPT2Attention_62405874811759
// MI455X (gfx1250) — matrix operand order repaired and checked by reading, not compiled, not run on hardware
//
#include <hip/hip_runtime.h>
#include <hip/hip_bf16.h>

// ---------------------------------------------------------------------------
// Types for CDNA5 WMMA (wave32): A/B = 16 bf16 per lane, C/D = 8 f32 per lane
// ---------------------------------------------------------------------------
typedef __attribute__((ext_vector_type(16))) __bf16 bf16x16;
typedef __attribute__((ext_vector_type(8)))  __bf16 bf16x8;
typedef __attribute__((ext_vector_type(8)))  float  f32x8;
typedef __attribute__((ext_vector_type(4)))  unsigned int uint32x4;
typedef __attribute__((ext_vector_type(8)))  int int32x8;
typedef __attribute__((ext_vector_type(4)))  int int32x4;

union FragAB {
  bf16x16 v;
  bf16x8  h[2];
};

#define D_MODEL 1024
#define SEQ     2048
#define NBATCH  2
#define NHEADS  16
#define HDIM    64
#define MROWS   4096   // B*S

#if defined(__HIP_DEVICE_COMPILE__) && __has_builtin(__builtin_amdgcn_tensor_load_to_lds)
#define USE_TDM 1
#else
#define USE_TDM 0
#endif

#if defined(__HIP_DEVICE_COMPILE__) && __has_builtin(__builtin_amdgcn_permlane16)
#define USE_PERMLANE 1
#else
#define USE_PERMLANE 0
#endif

// Row-wide (16-lane) max reduction: xor-butterfly on the VALU via permlane16.
__device__ __forceinline__ float rowmax16(float x) {
#if USE_PERMLANE
  const unsigned lo[4] = {0x67452301u, 0x54761032u, 0x32107654u, 0xFEDCBA98u};
  const unsigned hi[4] = {0xEFCDAB89u, 0xDCFE98BAu, 0xBA98FEDCu, 0x76543210u};
#pragma unroll
  for (int i = 0; i < 4; ++i) {
    unsigned xi = __float_as_uint(x);
    unsigned yi = __builtin_amdgcn_permlane16(xi, xi, lo[i], hi[i], false, false);
    x = fmaxf(x, __uint_as_float(yi));
  }
#else
#pragma unroll
  for (int msk = 8; msk >= 1; msk >>= 1)
    x = fmaxf(x, __shfl_xor(x, msk, 32));
#endif
  return x;
}

#if USE_TDM
// Issue a 2D tile DMA (global -> LDS) on the Tensor Data Mover.  D# per
// CDNA5 ISA ch.8: group0 = {count, lds_addr, global_addr, type=2};
// group1 = {data_size=2B, tensor dims, tile dims, dim0 stride}; groups 2/3
// zero (<=2D tensor).  Tracked by TENSORcnt.  6-arg builtin form
// (amdgpu-toolchain / clang-23).
__device__ __forceinline__ void tdm_load_2d(
    unsigned lds_addr, unsigned long long gaddr,
    unsigned tile_d0, unsigned tile_d1,
    unsigned tensor_d0, unsigned tensor_d1, unsigned stride_d0)
{
  uint32x4 g0;
  g0[0] = 1u;                                            // count=1 (valid user D#)
  g0[1] = lds_addr;                                      // LDS byte address
  g0[2] = (unsigned)(gaddr & 0xFFFFFFFFull);             // global_addr[31:0]
  g0[3] = (unsigned)((gaddr >> 32) & 0x1FFFFFFull)       // global_addr[56:32]
          | (2u << 30);                                  // type=2 ("image")
  int32x8 g1;
  g1[0] = (int)(1u << 16);                               // data_size=1 -> 2 bytes
  g1[1] = (int)(tensor_d0 << 16);                        // tensor_dim0[15:0]
  g1[2] = (int)((tensor_d0 >> 16) | (tensor_d1 << 16));  // dim0[31:16] | dim1[15:0]
  g1[3] = (int)((tensor_d1 >> 16) | (tile_d0 << 16));    // dim1[31:16] | tile_dim0
  g1[4] = (int)(tile_d1);                                // tile_dim1 (tile_dim2=0)
  g1[5] = (int)(stride_d0);                              // tensor_dim0_stride[31:0]
  g1[6] = 0;
  g1[7] = 0;
  int32x4 z4;
  z4[0] = 0; z4[1] = 0; z4[2] = 0; z4[3] = 0;
  int32x8 z8;
#pragma unroll
  for (int i = 0; i < 8; ++i) z8[i] = 0;
  __builtin_amdgcn_tensor_load_to_lds(g0, g1, z4, z4, z8, 0);
}
#endif

// ---------------------------------------------------------------------------
// GEMM: out[M,N] = A[M,K] * W[K,N] + bias[N]
//   A_BF16:   A operand is bf16 (staged to LDS via async-copy) else fp32.
//   OUT_MODE: 0 = fp32 row-major, 1 = bf16 row-major,
//             2 = bf16 transposed-per-head: Vt[(b*H+h)*64+dh][s]
// ---------------------------------------------------------------------------
template <bool A_BF16, int OUT_MODE>
__global__ __launch_bounds__(256) void gemm_bias_wmma(
    const void* __restrict__ Aip, const float* __restrict__ Bw,
    const float* __restrict__ bias, void* __restrict__ outp)
{
  __shared__ alignas(16) __bf16 As[64 * 40];    // [m][k], stride 40
  __shared__ alignas(16) __bf16 Bs[128 * 40];   // [n][k] (transposed), stride 40

  const int t    = threadIdx.x;
  const int lane = t & 31;
  const int w    = t >> 5;
  const int wm   = w & 1;
  const int wn   = w >> 1;
  const int r    = lane & 15;
  const int hh   = lane >> 4;

  const int M0 = blockIdx.y * 64;
  const int N0 = blockIdx.x * 128;

  f32x8 acc[2][2];
#pragma unroll
  for (int mi = 0; mi < 2; ++mi)
#pragma unroll
    for (int ni = 0; ni < 2; ++ni)
#pragma unroll
      for (int v = 0; v < 8; ++v) acc[mi][ni][v] = 0.0f;

  const int arow = t >> 2;
  const int acol = (t & 3) * 8;
  const int bk   = t >> 3;
  const int bn0  = (t & 7) * 16;

  for (int kk = 0; kk < D_MODEL; kk += 32) {
    if constexpr (A_BF16) {
      // bf16 source: 16B/lane copy on the CDNA5 async path (ASYNCcnt)
      const __bf16* src =
          (const __bf16*)Aip + (size_t)(M0 + arow) * D_MODEL + kk + acol;
      const unsigned lds_off = (unsigned)(uintptr_t)&As[arow * 40 + acol];
      asm volatile("global_load_async_to_lds_b128 %0, %1, off"
                   :: "v"(lds_off), "v"((unsigned long long)(uintptr_t)src)
                   : "memory");
    } else {
      const float* src = (const float*)Aip + (size_t)(M0 + arow) * D_MODEL + kk + acol;
      float4 f0 = reinterpret_cast<const float4*>(src)[0];
      float4 f1 = reinterpret_cast<const float4*>(src)[1];
      alignas(16) __bf16 tmp[8];
      tmp[0] = (__bf16)f0.x; tmp[1] = (__bf16)f0.y; tmp[2] = (__bf16)f0.z; tmp[3] = (__bf16)f0.w;
      tmp[4] = (__bf16)f1.x; tmp[5] = (__bf16)f1.y; tmp[6] = (__bf16)f1.z; tmp[7] = (__bf16)f1.w;
      *reinterpret_cast<uint4*>(&As[arow * 40 + acol]) = *reinterpret_cast<const uint4*>(tmp);
    }
    {
      const float* src = Bw + (size_t)(kk + bk) * D_MODEL + N0 + bn0;
#pragma unroll
      for (int j = 0; j < 16; ++j)
        Bs[(bn0 + j) * 40 + bk] = (__bf16)src[j];
    }
    if constexpr (A_BF16)
      asm volatile("s_wait_asynccnt 0" ::: "memory");
    __syncthreads();

    FragAB af[2], bfr[2];
#pragma unroll
    for (int mi = 0; mi < 2; ++mi) {
      const __bf16* p = &As[(wm * 32 + mi * 16 + r) * 40 + hh * 8];
      af[mi].h[0] = *reinterpret_cast<const bf16x8*>(p);
      af[mi].h[1] = *reinterpret_cast<const bf16x8*>(p + 16);
    }
#pragma unroll
    for (int ni = 0; ni < 2; ++ni) {
      const __bf16* p = &Bs[(wn * 32 + ni * 16 + r) * 40 + hh * 8];
      bfr[ni].h[0] = *reinterpret_cast<const bf16x8*>(p);
      bfr[ni].h[1] = *reinterpret_cast<const bf16x8*>(p + 16);
    }
#pragma unroll
    for (int mi = 0; mi < 2; ++mi)
#pragma unroll
      for (int ni = 0; ni < 2; ++ni)
        acc[mi][ni] = __builtin_amdgcn_wmma_f32_16x16x32_bf16(
            false, af[mi].v, false, bfr[ni].v, (short)0, acc[mi][ni], false, false);
    __syncthreads();
  }

#pragma unroll
  for (int mi = 0; mi < 2; ++mi)
#pragma unroll
    for (int ni = 0; ni < 2; ++ni) {
      const int gc = N0 + wn * 32 + ni * 16 + r;
      const float bb = bias[gc];
#pragma unroll
      for (int v = 0; v < 8; ++v) {
        const int gr = M0 + wm * 32 + mi * 16 + v + 8 * hh;
        const float val = acc[mi][ni][v] + bb;
        if constexpr (OUT_MODE == 0) {
          ((float*)outp)[(size_t)gr * D_MODEL + gc] = val;
        } else if constexpr (OUT_MODE == 1) {
          ((__bf16*)outp)[(size_t)gr * D_MODEL + gc] = (__bf16)val;
        } else {
          const size_t idx = ((size_t)(gr >> 11) * D_MODEL + gc) * SEQ + (gr & (SEQ - 1));
          ((__bf16*)outp)[idx] = (__bf16)val;
        }
      }
    }
}

// ---------------------------------------------------------------------------
// Flash attention: one workgroup per (batch, head, 64 q-rows).
// 128 threads = 4 waves; each wave owns 16 q-rows; key blocks of 32.
// K/V tiles staged to LDS by the Tensor Data Mover (double-buffered, one
// issuing wave, TENSORcnt), scores/PV on WMMA, softmax row-max on permlane16,
// row-sums on the matrix pipe (P x ones).
// ---------------------------------------------------------------------------
__global__ __launch_bounds__(128) void flash_attn_wmma(
    const __bf16* __restrict__ Qw,
    const __bf16* __restrict__ Kw,
    const __bf16* __restrict__ Vt,   // [b*1024 + head*64 + dh][s]
    __bf16* __restrict__ Ow)
{
  __shared__ alignas(16) __bf16 Ks[2][32 * 64];   // [key][dh]
  __shared__ alignas(16) __bf16 Vs[2][64 * 32];   // [dh][key]
  __shared__ alignas(16) __bf16 Pb[4][16 * 40];   // per-wave P tile [row][key]

  const int t    = threadIdx.x;
  const int lane = t & 31;
  const int w    = t >> 5;
  const int r    = lane & 15;
  const int hh   = lane >> 4;

  const int head = blockIdx.y;
  const int b    = blockIdx.z;
  const int q0   = blockIdx.x * 64 + w * 16;

  const size_t base  = (size_t)b * SEQ * D_MODEL + head * HDIM;
  const __bf16* kbp  = Kw + base;
  const __bf16* vtp  = Vt + ((size_t)b * D_MODEL + head * HDIM) * SEQ;

  // Q fragments (A layout): dh chunks c=0,1 straight from global
  FragAB qf[2];
#pragma unroll
  for (int c = 0; c < 2; ++c) {
    const __bf16* p = Qw + base + (size_t)(q0 + r) * D_MODEL + c * 32 + hh * 8;
    qf[c].h[0] = *reinterpret_cast<const bf16x8*>(p);
    qf[c].h[1] = *reinterpret_cast<const bf16x8*>(p + 16);
  }

  FragAB ones;
#pragma unroll
  for (int i = 0; i < 16; ++i) ones.v[i] = (__bf16)1.0f;

  f32x8 o[4];
#pragma unroll
  for (int nb = 0; nb < 4; ++nb)
#pragma unroll
    for (int v = 0; v < 8; ++v) o[nb][v] = 0.0f;

  float mrun[8], lrun[8], alpha[8];
#pragma unroll
  for (int v = 0; v < 8; ++v) { mrun[v] = -3.0e38f; lrun[v] = 0.0f; }

  // ---- K/V tile staging: TDM double-buffer (fallback: cooperative copy) ----
  auto stage = [&](int kb, int buf) {
#if USE_TDM
    if (w == 0) {
      tdm_load_2d((unsigned)(uintptr_t)&Ks[buf][0],
                  (unsigned long long)(uintptr_t)(kbp + (size_t)kb * D_MODEL),
                  /*tile_d0=*/HDIM, /*tile_d1=*/32,
                  /*tensor_d0=*/D_MODEL, /*tensor_d1=*/(1u << 20),
                  /*stride_d0=*/D_MODEL);
      tdm_load_2d((unsigned)(uintptr_t)&Vs[buf][0],
                  (unsigned long long)(uintptr_t)(vtp + kb),
                  /*tile_d0=*/32, /*tile_d1=*/HDIM,
                  /*tensor_d0=*/SEQ, /*tensor_d1=*/(1u << 20),
                  /*stride_d0=*/SEQ);
    }
#else
    {  // K tile: 32 keys x 64 dh (16 elems / thread)
      const int key = t >> 2, d0 = (t & 3) * 16;
      const __bf16* src = kbp + (size_t)(kb + key) * D_MODEL + d0;
      reinterpret_cast<uint4*>(&Ks[buf][key * 64 + d0])[0] =
          reinterpret_cast<const uint4*>(src)[0];
      reinterpret_cast<uint4*>(&Ks[buf][key * 64 + d0])[1] =
          reinterpret_cast<const uint4*>(src)[1];
    }
    {  // V tile: 64 dh x 32 keys
      const int dr = t >> 1, k0 = (t & 1) * 16;
      const __bf16* src = vtp + (size_t)dr * SEQ + kb + k0;
      reinterpret_cast<uint4*>(&Vs[buf][dr * 32 + k0])[0] =
          reinterpret_cast<const uint4*>(src)[0];
    }
#endif
  };

  stage(0, 0);

  const int NIB = SEQ / 32;
  for (int ib = 0; ib < NIB; ++ib) {
    const int cur = ib & 1;
#if USE_TDM
    if (w == 0) __builtin_amdgcn_s_wait_tensorcnt(0);
#endif
    __syncthreads();                       // cur buffers ready for all waves
    if (ib + 1 < NIB) stage((ib + 1) * 32, cur ^ 1);

    // ---- scores: two 16-key subblocks, accumulate over dh chunks ----
    f32x8 sc[2];
#pragma unroll
    for (int s = 0; s < 2; ++s) {
      f32x8 z;
#pragma unroll
      for (int v = 0; v < 8; ++v) z[v] = 0.0f;
#pragma unroll
      for (int c = 0; c < 2; ++c) {
        FragAB kf;  // B layout: lane = key column, dh runs of 8 at hh*8 and 16+hh*8
        const __bf16* p = &Ks[cur][(s * 16 + r) * 64 + c * 32 + hh * 8];
        kf.h[0] = *reinterpret_cast<const bf16x8*>(p);
        kf.h[1] = *reinterpret_cast<const bf16x8*>(p + 16);
        z = __builtin_amdgcn_wmma_f32_16x16x32_bf16(
            false, qf[c].v, false, kf.v, (short)0, z, false, false);
      }
      sc[s] = z;
    }

    // ---- online softmax: row max on VALU cross-lane permutes ----
#pragma unroll
    for (int v = 0; v < 8; ++v) {
      const float s0 = sc[0][v] * 0.125f;   // 1/sqrt(64)
      const float s1 = sc[1][v] * 0.125f;
      const float tmax = rowmax16(fmaxf(s0, s1));
      const float mnew = fmaxf(mrun[v], tmax);
      alpha[v] = __expf(mrun[v] - mnew);
      mrun[v]  = mnew;
      const float p0 = __expf(s0 - mnew);
      const float p1 = __expf(s1 - mnew);
      const int prow = v + 8 * hh;
      Pb[w][prow * 40 + r]      = (__bf16)p0;
      Pb[w][prow * 40 + 16 + r] = (__bf16)p1;
#pragma unroll
      for (int nb = 0; nb < 4; ++nb) o[nb][v] *= alpha[v];
    }

    // ---- P fragment (A layout) from per-wave LDS ----
    FragAB pf;
    {
      const __bf16* pp = &Pb[w][r * 40 + hh * 8];
      pf.h[0] = *reinterpret_cast<const bf16x8*>(pp);
      pf.h[1] = *reinterpret_cast<const bf16x8*>(pp + 16);
    }

    // ---- row sums on the matrix pipe: sums = P x ones ----
    {
      f32x8 z;
#pragma unroll
      for (int v = 0; v < 8; ++v) z[v] = 0.0f;
      f32x8 sums = __builtin_amdgcn_wmma_f32_16x16x32_bf16(
          false, pf.v, false, ones.v, (short)0, z, false, false);
#pragma unroll
      for (int v = 0; v < 8; ++v) lrun[v] = lrun[v] * alpha[v] + sums[v];
    }

    // ---- O += P * V from staged V^T tile ----
#pragma unroll
    for (int nb = 0; nb < 4; ++nb) {
      FragAB vf;
      const __bf16* vp = &Vs[cur][(nb * 16 + r) * 32 + hh * 8];
      vf.h[0] = *reinterpret_cast<const bf16x8*>(vp);
      vf.h[1] = *reinterpret_cast<const bf16x8*>(vp + 16);
      o[nb] = __builtin_amdgcn_wmma_f32_16x16x32_bf16(
          false, pf.v, false, vf.v, (short)0, o[nb], false, false);
    }
  }

  // ---- normalize and store bf16 attention output (row-major) ----
#pragma unroll
  for (int nb = 0; nb < 4; ++nb)
#pragma unroll
    for (int v = 0; v < 8; ++v) {
      const float val = o[nb][v] / lrun[v];
      const int row = q0 + v + 8 * hh;
      Ow[base + (size_t)row * D_MODEL + nb * 16 + r] = (__bf16)val;
    }
}

// ---------------------------------------------------------------------------
// Launch
// ---------------------------------------------------------------------------
extern "C" void kernel_launch(void* const* d_in, const int* in_sizes, int n_in,
                              void* d_out, int out_size, void* d_ws, size_t ws_size,
                              hipStream_t stream) {
  (void)in_sizes; (void)n_in; (void)out_size; (void)ws_size;

  const float* hq = (const float*)d_in[0];
  const float* hk = (const float*)d_in[1];
  const float* hv = (const float*)d_in[2];
  const float* Wq = (const float*)d_in[3];
  const float* bq = (const float*)d_in[4];
  const float* Wk = (const float*)d_in[5];
  const float* bk = (const float*)d_in[6];
  const float* Wv = (const float*)d_in[7];
  const float* bv = (const float*)d_in[8];
  const float* Wp = (const float*)d_in[9];
  const float* bp = (const float*)d_in[10];

  __bf16* ws = (__bf16*)d_ws;
  const size_t tsz = (size_t)MROWS * D_MODEL;
  __bf16* Qw = ws;
  __bf16* Kw = ws + tsz;
  __bf16* Vt = ws + 2 * tsz;                    // transposed-per-head layout
  __bf16* Aw = ws + 3 * tsz;

  dim3 gg(D_MODEL / 128, MROWS / 64);           // (8, 64)
  gemm_bias_wmma<false, 1><<<gg, 256, 0, stream>>>(hq, Wq, bq, Qw);
  gemm_bias_wmma<false, 1><<<gg, 256, 0, stream>>>(hk, Wk, bk, Kw);
  gemm_bias_wmma<false, 2><<<gg, 256, 0, stream>>>(hv, Wv, bv, Vt);

  dim3 ga(SEQ / 64, NHEADS, NBATCH);            // (32, 16, 2)
  flash_attn_wmma<<<ga, 128, 0, stream>>>(Qw, Kw, Vt, Aw);

  gemm_bias_wmma<true, 0><<<gg, 256, 0, stream>>>(Aw, Wp, bp, (float*)d_out);
}
